// DiscreteCosineTransformEmbedding_4045859193625
// MI455X (gfx1250) — matrix operand order repaired and checked by reading, not compiled, not run on hardware
//
#include <hip/hip_runtime.h>

// ---------------------------------------------------------------------------
// DCT-II along axis 1 of x:(B,100,32,3)  ==  per-batch GEMM  out_b = C @ x_b
//   C: 100x100 cosine basis (constant, synthesized on device)
//   x_b, out_b: 100x96 row-major f32
// Strategy: bf16 split (hi/lo) WMMA with f32 accumulation (fp32-class accuracy,
// 8x instruction density of fp32 WMMA) -> HBM-bound.
// x_b staged to LDS via GLOBAL_LOAD_ASYNC_TO_LDS_B128 (ASYNCcnt path).
// ---------------------------------------------------------------------------

typedef __attribute__((ext_vector_type(16))) __bf16        v16bf;
typedef __attribute__((ext_vector_type(8)))  float         v8f;
typedef __attribute__((ext_vector_type(8)))  unsigned int  v8u;

#define N_DCT 100          // true M and K
#define NCOL  96           // true N (32*3)
#define MT    7            // m-tiles of 16   (112 >= 100)
#define KS    4            // k-steps of 32   (128 >= 100)
#define NWAVE 6            // one 16-col n-tile per wave (6*16 = 96)
#define AFRAG_DWORDS (MT * KS * 32 * 8)   // 7168 dwords per (hi|lo) plane

union BF16Frag { v8u u; v16bf b; };

__device__ __forceinline__ unsigned short bf16_rne(float f) {
  unsigned u = __float_as_uint(f);
  unsigned r = u + 0x7FFFu + ((u >> 16) & 1u);   // round-to-nearest-even
  return (unsigned short)(r >> 16);
}

// A-matrix (16-bit, 16x32) per-lane K index, CDNA5 ISA "16-bit A 16x32":
//   lanes 0-15 : v0-3 -> K = 2v,2v+1        ; v4-7 -> K = 16+2(v-4), ...
//   lanes 16-31: v0-3 -> K = 8+2v, ...      ; v4-7 -> K = 24+2(v-4), ...
__device__ __forceinline__ int a_klocal(int lane, int v) {
  if (lane < 16) return (v < 4) ? (2 * v)     : (16 + 2 * (v - 4));
  else           return (v < 4) ? (8 + 2 * v) : (24 + 2 * (v - 4));
}

// ---------------------------------------------------------------------------
// Kernel 1: synthesize DCT matrix, split into bf16 hi/lo, store in exact
// per-lane WMMA A-fragment order:  ws[((mt*KS+ks)*32+lane)*8 + v]
//   hi plane at ws[0..7167], lo plane at ws[7168..14335]
// Padding rows (m>=100) and K (k>=100) are exact zeros.
// ---------------------------------------------------------------------------
__global__ void build_dct_afrags(unsigned int* __restrict__ ws) {
  int idx = blockIdx.x * blockDim.x + threadIdx.x;
  if (idx >= AFRAG_DWORDS) return;
  int v    = idx & 7;
  int lane = (idx >> 3) & 31;
  int ks   = (idx >> 8) & 3;
  int mt   = idx >> 10;

  int m  = mt * 16 + (lane & 15);
  int k0 = ks * 32 + a_klocal(lane, v);

  const double PI = 3.14159265358979323846;
  double f0 = 0.0, f1 = 0.0;
  if (m < N_DCT) {
    if (k0     < N_DCT) f0 = 2.0 * cos(PI * (2.0 * k0 + 1.0) * m / (2.0 * N_DCT));
    if (k0 + 1 < N_DCT) f1 = 2.0 * cos(PI * (2.0 * k0 + 3.0) * m / (2.0 * N_DCT));
  }
  float x0 = (float)f0, x1 = (float)f1;
  unsigned short h0 = bf16_rne(x0), h1 = bf16_rne(x1);
  float r0 = x0 - __uint_as_float((unsigned)h0 << 16);
  float r1 = x1 - __uint_as_float((unsigned)h1 << 16);
  ws[idx]                = (unsigned)h0 | ((unsigned)h1 << 16);
  ws[AFRAG_DWORDS + idx] = (unsigned)bf16_rne(r0) | ((unsigned)bf16_rne(r1) << 16);
}

// ---------------------------------------------------------------------------
// Kernel 2: one workgroup per batch b, 6 waves = 6 n-tiles.
// Async-stage x_b in LDS (memory->LDS direct path, ASYNCcnt), build B
// fragments with hi/lo bf16 split, read pre-swizzled A fragments from
// L2-resident workspace, 84 WMMAs/wave.
// ---------------------------------------------------------------------------
__global__ __launch_bounds__(NWAVE * 32) void dct_wmma_kernel(
    const float* __restrict__ x, float* __restrict__ out,
    const unsigned int* __restrict__ afrag) {
  __shared__ __align__(16) float xs[N_DCT * NCOL];   // 38400 B

  const int  tid  = threadIdx.x;
  const long base = (long)blockIdx.x * (N_DCT * NCOL);

  // Warm A-fragment table toward this WGP (L2-resident, reused by all blocks).
  __builtin_prefetch(afrag, 0, 3);

  // ---- stage x_b -> LDS via async direct-to-LDS b128 copies -------------
  // Generic address of a __shared__ object is {aperture_hi, lds_offset32};
  // the low 32 bits are exactly the wave-relative LDS byte address the
  // async op's VDST expects (ISA 10.2 aperture rules).
  const unsigned lds_base = (unsigned)(unsigned long long)(void*)xs;
  const float4*  gx       = (const float4*)(x + base);
  for (int i = tid; i < (N_DCT * NCOL) / 4; i += NWAVE * 32) {
    asm volatile("global_load_async_to_lds_b128 %0, %1, off"
                 :
                 : "v"(lds_base + (unsigned)(i * 16)), "v"(gx + i)
                 : "memory");
  }
#if __has_builtin(__builtin_amdgcn_s_wait_asynccnt)
  __builtin_amdgcn_s_wait_asynccnt(0);
#else
  asm volatile("s_wait_asynccnt 0" ::: "memory");
#endif
  __syncthreads();

  const int wave  = tid >> 5;          // n-tile index 0..5
  const int lane  = tid & 31;
  const int col   = wave * 16 + (lane & 15);
  const int khalf = (lane < 16) ? 0 : 16;   // unused; B K order follows a_klocal

  v8f acc[MT];
#pragma unroll
  for (int mt = 0; mt < MT; ++mt)
#pragma unroll
    for (int j = 0; j < 8; ++j) acc[mt][j] = 0.0f;

#pragma unroll
  for (int ks = 0; ks < KS; ++ks) {
    // ---- build B fragment (32x16 bf16): lane holds column `col`,
    //      vgpr v holds K = ks*32 + a_klocal(lane, v) + {0, 1}  (hi and lo planes)
    BF16Frag bhi, blo;
#pragma unroll
    for (int v = 0; v < 8; ++v) {
      int k0  = ks * 32 + a_klocal(lane, v);
      int kc0 = (k0 < N_DCT - 1) ? k0 : (N_DCT - 1);
      float e0 = xs[kc0 * NCOL + col];
      float e1 = xs[(kc0 + 1) * NCOL + col];
      e0 = (k0     < N_DCT) ? e0 : 0.0f;     // K padding -> exact zero
      e1 = (k0 + 1 < N_DCT) ? e1 : 0.0f;
      unsigned short h0 = bf16_rne(e0), h1 = bf16_rne(e1);
      float r0 = e0 - __uint_as_float((unsigned)h0 << 16);
      float r1 = e1 - __uint_as_float((unsigned)h1 << 16);
      bhi.u[v] = (unsigned)h0 | ((unsigned)h1 << 16);
      blo.u[v] = (unsigned)bf16_rne(r0) | ((unsigned)bf16_rne(r1) << 16);
    }

    // ---- 7 m-tiles: A fragments from L2-resident pre-swizzled workspace
#pragma unroll
    for (int mt = 0; mt < MT; ++mt) {
      int ab = ((mt * KS + ks) * 32 + lane) * 8;   // 32B/lane, coalesced
      BF16Frag ahi, alo;
      ahi.u = *(const v8u*)(afrag + ab);
      alo.u = *(const v8u*)(afrag + AFRAG_DWORDS + ab);
      // split-bf16: hi*hi + hi*lo + lo*hi, f32 accumulate
      acc[mt] = __builtin_amdgcn_wmma_f32_16x16x32_bf16(
          false, ahi.b, false, bhi.b, (short)0, acc[mt], false, false);
      acc[mt] = __builtin_amdgcn_wmma_f32_16x16x32_bf16(
          false, ahi.b, false, blo.b, (short)0, acc[mt], false, false);
      acc[mt] = __builtin_amdgcn_wmma_f32_16x16x32_bf16(
          false, alo.b, false, bhi.b, (short)0, acc[mt], false, false);
    }
  }

  // ---- store: f32 C/D layout: vgpr j -> row j (lanes 0-15) / j+8 (16-31)
  const int rbase = (lane < 16) ? 0 : 8;
#pragma unroll
  for (int mt = 0; mt < MT; ++mt)
#pragma unroll
    for (int j = 0; j < 8; ++j) {
      int row = mt * 16 + j + rbase;
      if (row < N_DCT)
        __builtin_nontemporal_store(acc[mt][j], out + base + row * NCOL + col);
    }
}

// ---------------------------------------------------------------------------
extern "C" void kernel_launch(void* const* d_in, const int* in_sizes, int n_in,
                              void* d_out, int out_size, void* d_ws, size_t ws_size,
                              hipStream_t stream) {
  const float* x   = (const float*)d_in[0];
  float*       out = (float*)d_out;
  unsigned*    ws  = (unsigned*)d_ws;

  const int nbatch = in_sizes[0] / (N_DCT * NCOL);   // 8192

  build_dct_afrags<<<(AFRAG_DWORDS + 255) / 256, 256, 0, stream>>>(ws);
  dct_wmma_kernel<<<nbatch, NWAVE * 32, 0, stream>>>(x, out, ws);
}
